// LiquidExecutor_45260365365713
// MI455X (gfx1250) — compile-verified, matrix operand order checked by reading, not run on hardware
//
#include <hip/hip_runtime.h>
#include <hip/hip_fp16.h>
#include <math.h>

// Problem constants (must match reference)
#define B_SZ 64
#define T_SZ 2048
#define DIN  64
#define H_SZ 256
#define O_SZ 32

typedef __attribute__((ext_vector_type(16))) _Float16 v16h;
typedef __attribute__((ext_vector_type(8)))  float    v8f;
typedef __attribute__((ext_vector_type(4)))  unsigned int u32x4;

union AF { v16h v; u32x4 q[2]; };

// Fast transcendentals on the critical path (guarded: compile-safe either way)
#if __has_builtin(__builtin_amdgcn_tanhf)
__device__ __forceinline__ float fast_tanh(float x) { return __builtin_amdgcn_tanhf(x); }
#else
__device__ __forceinline__ float fast_tanh(float x) { return tanhf(x); }
#endif
#if __has_builtin(__builtin_amdgcn_rcpf)
__device__ __forceinline__ float fast_rcp(float x) { return __builtin_amdgcn_rcpf(x); }
#else
__device__ __forceinline__ float fast_rcp(float x) { return 1.f / x; }
#endif
__device__ __forceinline__ float fast_sigmoid(float x) {
  return fast_rcp(1.f + __expf(-x));
}

__device__ __forceinline__ v8f wmma_f16(v16h a, v16h b, v8f c) {
  // D = A(16x32 f16) x B(32x16 f16) + C(16x16 f32)
  return __builtin_amdgcn_wmma_f32_16x16x32_f16(false, a, false, b, (short)0, c, false, false);
}

// A-fragment from a row-major f16 buffer (LDS or global).
// CDNA5 A layout: lane = M (mod 16); per lane the 16 halves cover
// K in [kTile*32 + hf*8, +8) and [kTile*32 + 16 + hf*8, +8)  (hf = lane>>4)
__device__ __forceinline__ v16h load_afrag(const _Float16* z, int strideH, int m, int kTile, int hf) {
  const _Float16* p = z + m * strideH + kTile * 32 + hf * 8;
  AF f;
  f.q[0] = *(const u32x4*)(p);
  f.q[1] = *(const u32x4*)(p + 16);
  return f.v;
}

// B-fragment from pre-packed weights: [tile][lane][16 halves] contiguous.
// The offset is laundered through an empty asm so the loads cannot be proven
// loop-invariant across the 2048-step time loop (prevents LICM mass-hoist ->
// scratch spill; weights must stream from L2 every step instead).
__device__ __forceinline__ v16h load_bfrag(const _Float16* base, int tile, int lane) {
  int off = (tile * 32 + lane) * 16;
  asm volatile("" : "+v"(off));
  const _Float16* p = base + off;
  AF f;
  f.q[0] = *(const u32x4*)(p);
  f.q[1] = *(const u32x4*)(p + 8);
  return f.v;
}

// ---------------------------------------------------------------------------
// Weight pre-pack: W (units x K, row-major f32), optional addend W2 (Wta+Wtb
// fusion), optional byte mask (NCP wiring).  Output: f16 fragments in
// [n_tile][k_tile][lane][16] order matching load_bfrag / the WMMA B layout
// (B^T has the same shape as A, lane = output-unit index).
// ---------------------------------------------------------------------------
__global__ void pack_gate_kernel(const float* __restrict__ W,
                                 const float* __restrict__ W2,
                                 const unsigned char* __restrict__ mask,
                                 int units, int K, int nTiles, int kTiles,
                                 _Float16* __restrict__ dst) {
  int idx = blockIdx.x * 256 + threadIdx.x;
  int total = nTiles * kTiles * 512;
  if (idx >= total) return;
  int j    = idx & 15;
  int lane = (idx >> 4) & 31;
  int tile = idx >> 9;
  int kt   = tile % kTiles;
  int nt   = tile / kTiles;
  int hf   = lane >> 4;
  int row  = nt * 16 + (lane & 15);
  int kl   = j + (j >= 8 ? 8 : 0) + hf * 8;
  int col  = kt * 32 + kl;
  float v = 0.f;
  if (row < units && col < K) {
    v = W[(size_t)row * K + col];
    if (W2)   v += W2[(size_t)row * K + col];
    if (mask) v *= (mask[(size_t)row * K + col] ? 1.f : 0.f);
  }
  dst[idx] = (_Float16)v;
}

__global__ void add_bias_kernel(const float* __restrict__ a,
                                const float* __restrict__ b,
                                float* __restrict__ o, int n) {
  int i = blockIdx.x * 256 + threadIdx.x;
  if (i < n) o[i] = a[i] + b[i];
}

// ---------------------------------------------------------------------------
// CfC recurrence epilogue: h = f1 + sig(t+bt)*(f2 - f1), f* = tanh(ff*+b*)
// C/D layout: element (vgpr i, lane) -> M = i + 8*(lane>>4), N = lane&15
// ---------------------------------------------------------------------------
__device__ __forceinline__ void cfc_epi(
    v8f a1, v8f a2, v8f at, int nt, int mrow, int hf, int units,
    const float* __restrict__ bf1, const float* __restrict__ bf2,
    const float* __restrict__ btt,
    _Float16* zself, int selfStride, int selfOff,
    _Float16* znext, int nextStride) {
  int unit = nt * 16 + mrow;
  if (unit < units) {
    float vb1 = bf1[unit], vb2 = bf2[unit], vbt = btt[unit];
#pragma unroll
    for (int i = 0; i < 8; ++i) {
      int r = i + hf * 8;
      float f1 = fast_tanh(a1[i] + vb1);
      float f2 = fast_tanh(a2[i] + vb2);
      float s  = fast_sigmoid(at[i] + vbt);
      float h  = f1 + s * (f2 - f1);
      _Float16 hh = (_Float16)h;
      zself[r * selfStride + selfOff + unit] = hh;  // own hidden for t+1
      znext[r * nextStride + unit]           = hh;  // input of next layer
    }
  }
}

// Depth-1 software-pipelined 3-gate K-loop: fetch k+1's B fragments while the
// three WMMAs for k issue, so the wait before each WMMA covers loads issued a
// full iteration earlier (hides L2 latency on the sequential critical path).
__device__ __forceinline__ void gate_kloop(
    const _Float16* zin, int strideH, int mrow, int hf, int lane,
    const _Float16* __restrict__ wf1, const _Float16* __restrict__ wf2,
    const _Float16* __restrict__ wtt, int tileBase, int kTiles,
    v8f& a1, v8f& a2, v8f& at) {
  v16h b1 = load_bfrag(wf1, tileBase, lane);
  v16h b2 = load_bfrag(wf2, tileBase, lane);
  v16h b3 = load_bfrag(wtt, tileBase, lane);
  for (int kt = 0; kt < kTiles; ++kt) {
    v16h av = load_afrag(zin, strideH, mrow, kt, hf);
    v16h n1 = b1, n2 = b2, n3 = b3;
    if (kt + 1 < kTiles) {
      n1 = load_bfrag(wf1, tileBase + kt + 1, lane);
      n2 = load_bfrag(wf2, tileBase + kt + 1, lane);
      n3 = load_bfrag(wtt, tileBase + kt + 1, lane);
    }
    a1 = wmma_f16(av, b1, a1);
    a2 = wmma_f16(av, b2, a2);
    at = wmma_f16(av, b3, at);
    b1 = n1; b2 = n2; b3 = n3;
  }
}

// ---------------------------------------------------------------------------
// Recurrent kernel: 4 workgroups (one per 16-batch slice), 8 wave32 each.
// Per timestep: fold-in input projection, then layers inter->command->motor.
// All activations in LDS; weights stream from L2 as packed B-fragments.
// ---------------------------------------------------------------------------
__global__ __launch_bounds__(256) void cfc_rnn_kernel(
    const float* __restrict__ x,
    const _Float16* __restrict__ Wp_p, const float* __restrict__ bp,
    const _Float16* __restrict__ g0f1, const _Float16* __restrict__ g0f2, const _Float16* __restrict__ g0tt,
    const float* __restrict__ b0f1, const float* __restrict__ b0f2, const float* __restrict__ b0tt,
    const _Float16* __restrict__ g1f1, const _Float16* __restrict__ g1f2, const _Float16* __restrict__ g1tt,
    const float* __restrict__ b1f1, const float* __restrict__ b1f2, const float* __restrict__ b1tt,
    const _Float16* __restrict__ g2f1, const _Float16* __restrict__ g2f2, const _Float16* __restrict__ g2tt,
    const float* __restrict__ b2f1, const float* __restrict__ b2f2, const float* __restrict__ b2tt,
    _Float16* __restrict__ cfc) {
  extern __shared__ __align__(16) char smem[];
  _Float16* z0 = (_Float16*)smem;       // 16 x 416  (proj 256 | h0 135 | pad)
  _Float16* z1 = z0 + 16 * 416;         // 16 x 224  (h0' 135 | h1 89)
  _Float16* z2 = z1 + 16 * 224;         // 16 x 128  (h1' 89  | h2 32 | pad)
  _Float16* xb = z2 + 16 * 128;         // 16 x 64   (x_t f16)

  const int tid  = threadIdx.x;
  const int lane = tid & 31;
  const int wv   = tid >> 5;
  const int hf   = lane >> 4;
  const int mrow = lane & 15;
  const int b0   = blockIdx.x * 16;

  // init hidden states + K-pad regions to zero
  for (int i = tid; i < 16 * 416; i += 256) if ((i % 416) >= 256) z0[i] = (_Float16)0.f;
  for (int i = tid; i < 16 * 224; i += 256) if ((i % 224) >= 135) z1[i] = (_Float16)0.f;
  for (int i = tid; i < 16 * 128; i += 256) if ((i % 128) >=  89) z2[i] = (_Float16)0.f;
  __syncthreads();

  for (int t = 0; t < T_SZ; ++t) {
    // ---- stage X: load x_t (16 x 64 f32 -> f16 LDS), prefetch next t
    {
      int e = tid * 4;
      int r = e >> 6, c = e & 63;
      const float* xp = x + (((size_t)(b0 + r)) * T_SZ + t) * DIN + c;
      float4 v4 = *(const float4*)xp;
      xb[r * 64 + c + 0] = (_Float16)v4.x;
      xb[r * 64 + c + 1] = (_Float16)v4.y;
      xb[r * 64 + c + 2] = (_Float16)v4.z;
      xb[r * 64 + c + 3] = (_Float16)v4.w;
      if (t + 1 < T_SZ) __builtin_prefetch(xp + DIN, 0, 1);
    }
    __syncthreads();

    // ---- stage P: input projection x_t @ Wp.T + bp -> z0[:,0:256]
    {
      v8f p0 = {}, p1 = {};
#pragma unroll
      for (int kt = 0; kt < 2; ++kt) {
        v16h av = load_afrag(xb, 64, mrow, kt, hf);
        p0 = wmma_f16(av, load_bfrag(Wp_p, (2 * wv + 0) * 2 + kt, lane), p0);
        p1 = wmma_f16(av, load_bfrag(Wp_p, (2 * wv + 1) * 2 + kt, lane), p1);
      }
      int u0 = (2 * wv + 0) * 16 + mrow;
      int u1 = (2 * wv + 1) * 16 + mrow;
      float bb0 = bp[u0], bb1 = bp[u1];
#pragma unroll
      for (int i = 0; i < 8; ++i) {
        int r = i + hf * 8;
        z0[r * 416 + u0] = (_Float16)(p0[i] + bb0);
        z0[r * 416 + u1] = (_Float16)(p1[i] + bb1);
      }
    }
    __syncthreads();

    // ---- stage 0: inter layer. z0 (16x416) -> 135 units. 9 N-tiles, 13 K-tiles.
    v8f a1 = {}, a2 = {}, at = {};
    gate_kloop(z0, 416, mrow, hf, lane, g0f1, g0f2, g0tt, wv * 13, 13, a1, a2, at);
    v8f c1 = {}, c2 = {}, ct = {};
    if (wv == 0)  // 9th N-tile
      gate_kloop(z0, 416, mrow, hf, lane, g0f1, g0f2, g0tt, 8 * 13, 13, c1, c2, ct);
    __syncthreads();  // all reads of z0 done before in-place h0 update
    cfc_epi(a1, a2, at, wv, mrow, hf, 135, b0f1, b0f2, b0tt, z0, 416, 256, z1, 224);
    if (wv == 0)
      cfc_epi(c1, c2, ct, 8, mrow, hf, 135, b0f1, b0f2, b0tt, z0, 416, 256, z1, 224);
    __syncthreads();

    // ---- stage 1: command layer. z1 (16x224) -> 89 units. 6 N-tiles, 7 K-tiles.
    v8f d1 = {}, d2 = {}, dt = {};
    if (wv < 6)
      gate_kloop(z1, 224, mrow, hf, lane, g1f1, g1f2, g1tt, wv * 7, 7, d1, d2, dt);
    __syncthreads();
    if (wv < 6)
      cfc_epi(d1, d2, dt, wv, mrow, hf, 89, b1f1, b1f2, b1tt, z1, 224, 135, z2, 128);
    __syncthreads();

    // ---- stage 2: motor layer. z2 (16x128) -> 32 units. 2 N-tiles, 4 K-tiles.
    v8f e1 = {}, e2 = {}, et = {};
    if (wv < 2)
      gate_kloop(z2, 128, mrow, hf, lane, g2f1, g2f2, g2tt, wv * 4, 4, e1, e2, et);
    __syncthreads();
    if (wv < 2) {
      int unit = wv * 16 + mrow;  // < 32 always
      float vb1 = b2f1[unit], vb2 = b2f2[unit], vbt = b2tt[unit];
#pragma unroll
      for (int i = 0; i < 8; ++i) {
        int r = i + hf * 8;
        float f1 = fast_tanh(e1[i] + vb1);
        float f2 = fast_tanh(e2[i] + vb2);
        float s  = fast_sigmoid(et[i] + vbt);
        float h  = f1 + s * (f2 - f1);
        _Float16 hh = (_Float16)h;
        z2[r * 128 + 89 + unit] = hh;  // h2 for t+1
        cfc[(((size_t)(b0 + r)) * T_SZ + t) * O_SZ + unit] = hh;  // motor output
      }
    }
    __syncthreads();
  }
}

// ---------------------------------------------------------------------------
// Head: out = gelu(cfc @ Wh1.T + bh1) @ Wh2.T + bh2, fully parallel over B*T.
// Each wave owns one 16-row M tile; gelu intermediate kept in a private LDS
// slice and re-read as A fragments for the second GEMM.
// ---------------------------------------------------------------------------
__global__ __launch_bounds__(256) void head_kernel(
    const _Float16* __restrict__ cfc,
    const _Float16* __restrict__ Wh1_p, const float* __restrict__ bh1,
    const _Float16* __restrict__ Wh2_p, const float* __restrict__ bh2,
    float* __restrict__ out) {
  extern __shared__ __align__(16) char smem[];
  const int tid = threadIdx.x, lane = tid & 31, wv = tid >> 5;
  const int hf = lane >> 4, mrow = lane & 15;
  _Float16* hb = (_Float16*)smem + (size_t)wv * 16 * 256;  // 8 KB per wave
  const int mtile = blockIdx.x * 8 + wv;

  v8f zero = {};
  v8f acc[16];
#pragma unroll
  for (int i = 0; i < 16; ++i) acc[i] = zero;

  // GEMM1: (16x32) x (32x256); K=32 is exactly one K-tile.
  v16h a0 = load_afrag(cfc + (size_t)mtile * 16 * 32, 32, mrow, 0, hf);
#pragma unroll
  for (int nt = 0; nt < 16; ++nt)
    acc[nt] = wmma_f16(a0, load_bfrag(Wh1_p, nt, lane), acc[nt]);

  // exact GELU, store f16 to LDS
#pragma unroll
  for (int nt = 0; nt < 16; ++nt) {
    int unit = nt * 16 + mrow;
    float bb = bh1[unit];
#pragma unroll
    for (int i = 0; i < 8; ++i) {
      int r = i + hf * 8;
      float v = acc[nt][i] + bb;
      float g = 0.5f * v * (1.f + erff(v * 0.70710678118654752f));
      hb[r * 256 + unit] = (_Float16)g;
    }
  }
  __syncthreads();  // (per-wave slice; barrier for ordering safety)

  // GEMM2: (16x256) x (256x32); 8 K-tiles, 2 N-tiles.
  v8f o0 = zero, o1 = zero;
#pragma unroll
  for (int kt = 0; kt < 8; ++kt) {
    v16h av = load_afrag(hb, 256, mrow, kt, hf);
    o0 = wmma_f16(av, load_bfrag(Wh2_p, 0 * 8 + kt, lane), o0);
    o1 = wmma_f16(av, load_bfrag(Wh2_p, 1 * 8 + kt, lane), o1);
  }
  float bo0 = bh2[mrow], bo1 = bh2[16 + mrow];
#pragma unroll
  for (int i = 0; i < 8; ++i) {
    size_t r = (size_t)mtile * 16 + i + hf * 8;
    out[r * 32 + mrow]      = o0[i] + bo0;
    out[r * 32 + 16 + mrow] = o1[i] + bo1;
  }
}

// ---------------------------------------------------------------------------
extern "C" void kernel_launch(void* const* d_in, const int* in_sizes, int n_in,
                              void* d_out, int out_size, void* d_ws, size_t ws_size,
                              hipStream_t stream) {
  (void)in_sizes; (void)n_in; (void)out_size; (void)ws_size;

  const float* x  = (const float*)d_in[0];
  const float* Wp = (const float*)d_in[1];
  const float* bp = (const float*)d_in[2];
  const float* Wff1[3] = {(const float*)d_in[3],  (const float*)d_in[12], (const float*)d_in[21]};
  const float* bff1[3] = {(const float*)d_in[4],  (const float*)d_in[13], (const float*)d_in[22]};
  const float* Wff2[3] = {(const float*)d_in[5],  (const float*)d_in[14], (const float*)d_in[23]};
  const float* bff2[3] = {(const float*)d_in[6],  (const float*)d_in[15], (const float*)d_in[24]};
  const float* Wta[3]  = {(const float*)d_in[7],  (const float*)d_in[16], (const float*)d_in[25]};
  const float* bta[3]  = {(const float*)d_in[8],  (const float*)d_in[17], (const float*)d_in[26]};
  const float* Wtb[3]  = {(const float*)d_in[9],  (const float*)d_in[18], (const float*)d_in[27]};
  const float* btb[3]  = {(const float*)d_in[10], (const float*)d_in[19], (const float*)d_in[28]};
  const unsigned char* mask[3] = {(const unsigned char*)d_in[11],
                                  (const unsigned char*)d_in[20],
                                  (const unsigned char*)d_in[29]};
  const float* Wh1 = (const float*)d_in[30];
  const float* bh1 = (const float*)d_in[31];
  const float* Wh2 = (const float*)d_in[32];
  const float* bh2 = (const float*)d_in[33];
  float* out = (float*)d_out;

  const int units[3]  = {135, 89, 32};
  const int Kdim[3]   = {391, 224, 121};
  const int nT[3]     = {9, 6, 2};
  const int kT[3]     = {13, 7, 4};

  size_t off = 0;
  auto carve = [&](size_t bytes) -> char* {
    off = (off + 255) & ~(size_t)255;
    char* p = (char*)d_ws + off;
    off += bytes;
    return p;
  };

  _Float16* cfc   = (_Float16*)carve((size_t)B_SZ * T_SZ * O_SZ * sizeof(_Float16));
  _Float16* Wp_p  = (_Float16*)carve((size_t)16 * 2 * 512 * sizeof(_Float16));
  _Float16* gf1[3]; _Float16* gf2[3]; _Float16* gtt[3]; float* btt[3];
  for (int l = 0; l < 3; ++l) {
    size_t bytes = (size_t)nT[l] * kT[l] * 512 * sizeof(_Float16);
    gf1[l] = (_Float16*)carve(bytes);
    gf2[l] = (_Float16*)carve(bytes);
    gtt[l] = (_Float16*)carve(bytes);
    btt[l] = (float*)carve((size_t)units[l] * sizeof(float));
  }
  _Float16* Wh1_p = (_Float16*)carve((size_t)16 * 1 * 512 * sizeof(_Float16));
  _Float16* Wh2_p = (_Float16*)carve((size_t)2 * 8 * 512 * sizeof(_Float16));

  // --- weight pre-pack (off critical path, parallel) ---
  auto launch_pack = [&](const float* W, const float* W2, const unsigned char* m,
                         int u, int K, int nt, int kt, _Float16* dst) {
    int total = nt * kt * 512;
    pack_gate_kernel<<<(total + 255) / 256, 256, 0, stream>>>(W, W2, m, u, K, nt, kt, dst);
  };
  launch_pack(Wp, nullptr, nullptr, H_SZ, DIN, 16, 2, Wp_p);
  for (int l = 0; l < 3; ++l) {
    launch_pack(Wff1[l], nullptr, mask[l], units[l], Kdim[l], nT[l], kT[l], gf1[l]);
    launch_pack(Wff2[l], nullptr, mask[l], units[l], Kdim[l], nT[l], kT[l], gf2[l]);
    launch_pack(Wta[l],  Wtb[l],  nullptr, units[l], Kdim[l], nT[l], kT[l], gtt[l]);
    add_bias_kernel<<<1, 256, 0, stream>>>(bta[l], btb[l], btt[l], units[l]);
  }
  launch_pack(Wh1, nullptr, nullptr, H_SZ, O_SZ, 16, 1, Wh1_p);
  launch_pack(Wh2, nullptr, nullptr, O_SZ, H_SZ, 2, 8, Wh2_p);

  // --- recurrent scan: 4 WGs x 8 waves, LDS-resident activations ---
  size_t rnnSmem = (size_t)(16 * 416 + 16 * 224 + 16 * 128 + 16 * 64) * sizeof(_Float16);
  cfc_rnn_kernel<<<B_SZ / 16, 256, rnnSmem, stream>>>(
      x, Wp_p, bp,
      gf1[0], gf2[0], gtt[0], bff1[0], bff2[0], btt[0],
      gf1[1], gf2[1], gtt[1], bff1[1], bff2[1], btt[1],
      gf1[2], gf2[2], gtt[2], bff1[2], bff2[2], btt[2],
      cfc);

  // --- GELU head: fully parallel over B*T rows ---
  size_t headSmem = (size_t)8 * 16 * 256 * sizeof(_Float16);
  head_kernel<<<(B_SZ * T_SZ) / (16 * 8), 256, headSmem, stream>>>(
      cfc, Wh1_p, bh1, Wh2_p, bh2, out);
}
